// MultiHeadAttention_20031727468654
// MI455X (gfx1250) — matrix operand order repaired and checked by reading, not compiled, not run on hardware
//
#include <hip/hip_runtime.h>
#include <hip/hip_bf16.h>

// ---------------------------------------------------------------------------
// MI455X (gfx1250) fused multi-head attention, bf16 WMMA + flash softmax.
// ---------------------------------------------------------------------------

typedef __bf16 v16bf __attribute__((ext_vector_type(16)));
typedef __bf16 v8bf  __attribute__((ext_vector_type(8)));
typedef float  v8f   __attribute__((ext_vector_type(8)));

#define WMMA_BF16(a, b, c) \
  __builtin_amdgcn_wmma_f32_16x16x32_bf16(false, (a), false, (b), (short)0, (c), false, false)

#define BATCH 4
#define SEQ   2048
#define CDIM  768
#define HEADS 12
#define DHEAD 64

// round-to-nearest-even f32 -> bf16 bits (pure integer ops, always selects)
static __device__ __forceinline__ unsigned short f32_to_bf16_bits(float f) {
  unsigned int u = __float_as_uint(f);
  u += 0x7FFFu + ((u >> 16) & 1u);
  return (unsigned short)(u >> 16);
}

// A-matrix fragment 16x32 bf16:
// lane m = l&15, half h = l>>4; element e -> K = h*16 + e.
// One contiguous 32B load per lane from a row-major source.
static __device__ __forceinline__ v16bf load_a_frag(const unsigned short* base,
                                                    int ld, int lane) {
  const int m = lane & 15, h = lane >> 4;
  return *(const v16bf*)(base + (size_t)m * ld + h * 16);
}

// B-matrix fragment 32x16 bf16: lane n = l&15, half h = l>>4;
// element e -> K = h*16 + e.  B[k][n] = src[n*ld + k]  (src rows are the
// output columns, so row-major W / K-matrix / V^T feed this directly with
// one contiguous 32B load per lane).
static __device__ __forceinline__ v16bf load_b_frag(const unsigned short* base,
                                                    int ld, int lane) {
  const int n = lane & 15, h = lane >> 4;
  return *(const v16bf*)(base + (size_t)n * ld + h * 16);
}

// ---------------------------------------------------------------------------
// 1) f32 -> bf16 conversion
// ---------------------------------------------------------------------------
__global__ void cvt_bf16_kernel(const float* __restrict__ in,
                                unsigned short* __restrict__ out, int n) {
  int i = blockIdx.x * 256 + threadIdx.x;
  if (i < n) out[i] = f32_to_bf16_bits(in[i]);
}

// ---------------------------------------------------------------------------
// 2) QKV projection: y = x @ W^T + b, wave tile 32x64, 8 wmma / K-step.
//    Q scaled by 1/sqrt(D) and stored [B,H,N,D]; K stored [B,H,N,D];
//    V stored transposed [B,H,D,N] so P*V B-frags are contiguous.
// ---------------------------------------------------------------------------
__global__ __launch_bounds__(32) void qkv_gemm_kernel(
    const unsigned short* __restrict__ xb,
    const unsigned short* __restrict__ wq, const unsigned short* __restrict__ wk,
    const unsigned short* __restrict__ wv,
    const float* __restrict__ bq, const float* __restrict__ bk,
    const float* __restrict__ bv,
    unsigned short* __restrict__ q, unsigned short* __restrict__ k,
    unsigned short* __restrict__ vt) {
  const int lane  = threadIdx.x;
  const int which = blockIdx.z;              // 0=Q 1=K 2=V (wave-uniform)
  const int mb    = blockIdx.x * 32;         // row tile in [0, B*N)
  const int nb    = blockIdx.y * 64;         // col tile in [0, C)
  const unsigned short* W = (which == 0) ? wq : (which == 1) ? wk : wv;
  const float* bias       = (which == 0) ? bq : (which == 1) ? bk : bv;

  v8f acc[2][4];
#pragma unroll
  for (int i = 0; i < 2; ++i)
#pragma unroll
    for (int j = 0; j < 4; ++j) acc[i][j] = (v8f){};

  for (int kk = 0; kk < CDIM; kk += 32) {
    v16bf a0 = load_a_frag(xb + (size_t)mb * CDIM + kk, CDIM, lane);
    v16bf a1 = load_a_frag(xb + (size_t)(mb + 16) * CDIM + kk, CDIM, lane);
#pragma unroll
    for (int j = 0; j < 4; ++j) {
      v16bf b = load_b_frag(W + (size_t)(nb + j * 16) * CDIM + kk, CDIM, lane);
      acc[0][j] = WMMA_BF16(a0, b, acc[0][j]);
      acc[1][j] = WMMA_BF16(a1, b, acc[1][j]);
    }
  }

  const float scale = (which == 0) ? 0.125f : 1.0f;  // D^-0.5 folded into Q
  const int h = lane >> 4, n16 = lane & 15;
#pragma unroll
  for (int mi = 0; mi < 2; ++mi)
#pragma unroll
    for (int j = 0; j < 4; ++j)
#pragma unroll
      for (int r = 0; r < 8; ++r) {
        int row = mb + mi * 16 + r + 8 * h;          // global token index
        int o   = nb + j * 16 + n16;                 // output feature
        float y = (acc[mi][j][r] + bias[o]) * scale;
        unsigned short val = f32_to_bf16_bits(y);
        int bb = row >> 11, nn = row & (SEQ - 1);
        int hd = o >> 6, d = o & (DHEAD - 1);
        if (which == 2)
          vt[(((size_t)bb * HEADS + hd) * DHEAD + d) * SEQ + nn] = val;
        else {
          unsigned short* dst = (which == 0) ? q : k;
          dst[(((size_t)bb * HEADS + hd) * SEQ + nn) * DHEAD + d] = val;
        }
      }
}

// ---------------------------------------------------------------------------
// 3) Flash attention: one wave = 16 query rows of one (b,h).
//    Per 32-key block: 4 wmma for S = Q K^T (D=64 in two k-chunks, two
//    16-col S tiles), online softmax (shfl_xor row reductions over the
//    16-lane halves), P -> LDS (C-layout) -> A-frag, 4 wmma for P V.
// ---------------------------------------------------------------------------
__global__ __launch_bounds__(256) void attn_kernel(
    const unsigned short* __restrict__ q,
    const unsigned short* __restrict__ kmat,
    const unsigned short* __restrict__ vt,
    unsigned short* __restrict__ ctx) {
  __shared__ __align__(32) unsigned short pbuf[8][16 * 32];  // per-wave P tile

  const int lane = threadIdx.x & 31;
  const int wv   = threadIdx.x >> 5;
  const int tile = blockIdx.x * 8 + wv;     // global wave id
  const int qb   = (tile & 127) * 16;       // query-row base within sequence
  const int bh   = tile >> 7;               // b*HEADS + head

  const unsigned short* Q = q + ((size_t)bh * SEQ + qb) * DHEAD;
  const unsigned short* K = kmat + (size_t)bh * SEQ * DHEAD;
  const unsigned short* V = vt + (size_t)bh * DHEAD * SEQ;

  const v16bf aq0 = load_a_frag(Q, DHEAD, lane);       // d = 0..31
  const v16bf aq1 = load_a_frag(Q + 32, DHEAD, lane);  // d = 32..63

  v8f acc[4];
#pragma unroll
  for (int j = 0; j < 4; ++j) acc[j] = (v8f){};
  float mrow[8], lrow[8];
#pragma unroll
  for (int r = 0; r < 8; ++r) { mrow[r] = -1e30f; lrow[r] = 0.0f; }

  const int h = lane >> 4, n16 = lane & 15;
  unsigned short* P = &pbuf[wv][0];

  for (int kb = 0; kb < SEQ; kb += 32) {
    v8f s0 = (v8f){}, s1 = (v8f){};
    {
      v16bf b0 = load_b_frag(K + (size_t)kb * DHEAD, DHEAD, lane);
      v16bf b1 = load_b_frag(K + (size_t)kb * DHEAD + 32, DHEAD, lane);
      s0 = WMMA_BF16(aq0, b0, s0);
      s0 = WMMA_BF16(aq1, b1, s0);
      v16bf b2 = load_b_frag(K + (size_t)(kb + 16) * DHEAD, DHEAD, lane);
      v16bf b3 = load_b_frag(K + (size_t)(kb + 16) * DHEAD + 32, DHEAD, lane);
      s1 = WMMA_BF16(aq0, b2, s1);
      s1 = WMMA_BF16(aq1, b3, s1);
    }
#pragma unroll
    for (int r = 0; r < 8; ++r) {
      float vmax = fmaxf(s0[r], s1[r]);
      vmax = fmaxf(vmax, __shfl_xor(vmax, 1, 32));
      vmax = fmaxf(vmax, __shfl_xor(vmax, 2, 32));
      vmax = fmaxf(vmax, __shfl_xor(vmax, 4, 32));
      vmax = fmaxf(vmax, __shfl_xor(vmax, 8, 32));   // row max, per 16-lane half
      float mnew = fmaxf(mrow[r], vmax);
      float corr = __expf(mrow[r] - mnew);
      float p0 = __expf(s0[r] - mnew);
      float p1 = __expf(s1[r] - mnew);
      float ps = p0 + p1;
      ps += __shfl_xor(ps, 1, 32);
      ps += __shfl_xor(ps, 2, 32);
      ps += __shfl_xor(ps, 4, 32);
      ps += __shfl_xor(ps, 8, 32);
      lrow[r] = lrow[r] * corr + ps;
      mrow[r] = mnew;
#pragma unroll
      for (int j = 0; j < 4; ++j) acc[j][r] *= corr;
      const int m = r + 8 * h;                       // C-layout row
      P[m * 32 + n16]      = f32_to_bf16_bits(p0);   // keys kb + 0..15
      P[m * 32 + 16 + n16] = f32_to_bf16_bits(p1);   // keys kb + 16..31
    }
    // wave-private LDS region; per-wave LDS ops are in-order -> no barrier
    v16bf pa = load_a_frag(P, 32, lane);             // P as 16x32 A fragment
#pragma unroll
    for (int j = 0; j < 4; ++j) {
      v16bf bv_ = load_b_frag(V + (size_t)(j * 16) * SEQ + kb, SEQ, lane);
      acc[j] = WMMA_BF16(pa, bv_, acc[j]);
    }
  }

  const int head = bh % HEADS, bb = bh / HEADS;
#pragma unroll
  for (int j = 0; j < 4; ++j)
#pragma unroll
    for (int r = 0; r < 8; ++r) {
      int row = qb + r + 8 * h;
      int d   = j * 16 + n16;
      float o = acc[j][r] / lrow[r];
      ctx[((size_t)bb * SEQ + row) * CDIM + head * DHEAD + d] =
          f32_to_bf16_bits(o);
    }
}

// ---------------------------------------------------------------------------
// 4) Output projection: out = ctx @ Wo^T + bo  (f32 output)
// ---------------------------------------------------------------------------
__global__ __launch_bounds__(32) void proj_gemm_kernel(
    const unsigned short* __restrict__ ctx,
    const unsigned short* __restrict__ wo,
    const float* __restrict__ bo, float* __restrict__ out) {
  const int lane = threadIdx.x;
  const int mb = blockIdx.x * 32;
  const int nb = blockIdx.y * 64;

  v8f acc[2][4];
#pragma unroll
  for (int i = 0; i < 2; ++i)
#pragma unroll
    for (int j = 0; j < 4; ++j) acc[i][j] = (v8f){};

  for (int kk = 0; kk < CDIM; kk += 32) {
    v16bf a0 = load_a_frag(ctx + (size_t)mb * CDIM + kk, CDIM, lane);
    v16bf a1 = load_a_frag(ctx + (size_t)(mb + 16) * CDIM + kk, CDIM, lane);
#pragma unroll
    for (int j = 0; j < 4; ++j) {
      v16bf b = load_b_frag(wo + (size_t)(nb + j * 16) * CDIM + kk, CDIM, lane);
      acc[0][j] = WMMA_BF16(a0, b, acc[0][j]);
      acc[1][j] = WMMA_BF16(a1, b, acc[1][j]);
    }
  }

  const int h = lane >> 4, n16 = lane & 15;
#pragma unroll
  for (int mi = 0; mi < 2; ++mi)
#pragma unroll
    for (int j = 0; j < 4; ++j)
#pragma unroll
      for (int r = 0; r < 8; ++r) {
        int row = mb + mi * 16 + r + 8 * h;
        int o   = nb + j * 16 + n16;
        out[(size_t)row * CDIM + o] = acc[mi][j][r] + bo[o];
      }
}

// ---------------------------------------------------------------------------
// Host-side orchestration
// ---------------------------------------------------------------------------
extern "C" void kernel_launch(void* const* d_in, const int* in_sizes, int n_in,
                              void* d_out, int out_size, void* d_ws,
                              size_t ws_size, hipStream_t stream) {
  const float* x  = (const float*)d_in[0];
  const float* Wq = (const float*)d_in[1];
  const float* bq = (const float*)d_in[2];
  const float* Wk = (const float*)d_in[3];
  const float* bk = (const float*)d_in[4];
  const float* Wv = (const float*)d_in[5];
  const float* bv = (const float*)d_in[6];
  const float* Wo = (const float*)d_in[7];
  const float* bo = (const float*)d_in[8];

  const size_t XN = (size_t)BATCH * SEQ * CDIM;  // 6,291,456
  const size_t WN = (size_t)CDIM * CDIM;         //   589,824

  char* ws = (char*)d_ws;
  size_t off = 0;
  auto carve = [&](size_t bytes) {
    void* p = ws + off;
    off = (off + bytes + 255) & ~(size_t)255;
    return p;
  };
  unsigned short* xb  = (unsigned short*)carve(XN * 2);
  unsigned short* wqb = (unsigned short*)carve(WN * 2);
  unsigned short* wkb = (unsigned short*)carve(WN * 2);
  unsigned short* wvb = (unsigned short*)carve(WN * 2);
  unsigned short* wob = (unsigned short*)carve(WN * 2);
  unsigned short* qB  = (unsigned short*)carve(XN * 2);  // [B,H,N,D]
  unsigned short* kB  = (unsigned short*)carve(XN * 2);  // [B,H,N,D]
  unsigned short* vtB = (unsigned short*)carve(XN * 2);  // [B,H,D,N]
  unsigned short* ctx = (unsigned short*)carve(XN * 2);  // [B,N,C]
  (void)ws_size; (void)in_sizes; (void)n_in; (void)out_size;

  // 1) convert inputs to bf16
  cvt_bf16_kernel<<<(XN + 255) / 256, 256, 0, stream>>>(x, xb, (int)XN);
  cvt_bf16_kernel<<<(WN + 255) / 256, 256, 0, stream>>>(Wq, wqb, (int)WN);
  cvt_bf16_kernel<<<(WN + 255) / 256, 256, 0, stream>>>(Wk, wkb, (int)WN);
  cvt_bf16_kernel<<<(WN + 255) / 256, 256, 0, stream>>>(Wv, wvb, (int)WN);
  cvt_bf16_kernel<<<(WN + 255) / 256, 256, 0, stream>>>(Wo, wob, (int)WN);

  // 2) QKV projections (z: 0=Q 1=K 2=V)
  {
    dim3 grid((BATCH * SEQ) / 32, CDIM / 64, 3);
    qkv_gemm_kernel<<<grid, 32, 0, stream>>>(xb, wqb, wkb, wvb, bq, bk, bv,
                                             qB, kB, vtB);
  }

  // 3) fused flash attention: B*H*(N/16) = 6144 waves, 8 waves/block
  attn_kernel<<<(BATCH * HEADS * (SEQ / 16)) / 8, 256, 0, stream>>>(qB, kB,
                                                                    vtB, ctx);

  // 4) output projection (f32 + bias)
  {
    dim3 grid((BATCH * SEQ) / 32, CDIM / 64);
    proj_gemm_kernel<<<grid, 32, 0, stream>>>(ctx, wob, bo, (float*)d_out);
  }
}
